// SpatialGraphConv_16269336117516
// MI455X (gfx1250) — matrix operand order repaired and checked by reading, not compiled, not run on hardware
//
#include <hip/hip_runtime.h>

// ---------------------------------------------------------------------------
// SpatialGraphConv (ST-GCN block) for MI455X / gfx1250, wave32 + WMMA bf16.
// out[n,c,t,w] = relu( sum_v (inv_c*gcn_w @ x)[c,t,v] * (A*edge)[v,w]
//                      + const_c + (rinv_c*res_w @ x)[c,t,w] )
// BN folded into weights/biases; both GEMMs run on v_wmma_f32_16x16x32_bf16
// with fp32 accumulation. Memory-bound target: ~370MB traffic @ 23.3 TB/s.
// Streaming x loads / out stores use non-temporal cache hints (TH=NT).
// ---------------------------------------------------------------------------

typedef __attribute__((ext_vector_type(16))) __bf16 v16bf;
typedef __attribute__((ext_vector_type(8)))  float  v8f;
typedef __attribute__((ext_vector_type(4)))  float  vf4;

union BF16Frag {
  v16bf v;
  uint4 q[2];
};

__device__ __forceinline__ unsigned short f2bf(float f) {
  unsigned int u = __float_as_uint(f);
  u += 0x7FFFu + ((u >> 16) & 1u);           // round-to-nearest-even
  return (unsigned short)(u >> 16);
}
__device__ __forceinline__ float bf2f(unsigned short h) {
  return __uint_as_float(((unsigned int)h) << 16);
}
__device__ __forceinline__ v8f wmma_bf16(BF16Frag a, BF16Frag b, v8f c) {
  return __builtin_amdgcn_wmma_f32_16x16x32_bf16(
      /*neg_a=*/false, a.v, /*neg_b=*/false, b.v,
      /*c_mod=*/(short)0, c, /*reuse_a=*/false, /*reuse_b=*/false);
}

// Problem constants
#define NB    64     // batch
#define CIN   64
#define COUT  128
#define TDIM  300
#define VDIM  25
#define TT    12     // t per workgroup (25 blocks * 12 = 300, no tail)
#define COLS  300    // TT*VDIM
#define NCT   19     // ceil(304/16) column tiles
#define XB_STRIDE 72 // bf16 stride per column (16B aligned, conflict-free)
#define YB_CSTRIDE 392 // 12*32 + 8 (bank-skewed)
#define RB_CSTRIDE 304

// LDS layout (ushort/float indices into one extern-shared block)
#define WCAT_US   0        // ushort[256*64]  folded weights (gcn|res)
#define AWBT_US   16384    // ushort[32*32]   (A*edge)^T, zero-padded
#define BIASA_F   8704     // float[256]      per-row GEMM bias
#define CONSTC_F  8960     // float[128]      BN shift sum
#define XB_US     18176    // ushort[304*72]  X tile, column-major bf16
#define YB_US     40064    // ushort[128*392] Y1 (gcn branch), [c][t][32]
#define RB_US     90240    // ushort[128*304] residual branch, [c][t*25+v]
#define SMEM_BYTES 258304

extern "C" __global__ void __launch_bounds__(256)
stgcn_fused_kernel(const float* __restrict__ x,
                   const float* __restrict__ A,
                   const float* __restrict__ edge,
                   const float* __restrict__ gcn_w,
                   const float* __restrict__ gcn_b,
                   const float* __restrict__ bn_gamma,
                   const float* __restrict__ bn_beta,
                   const float* __restrict__ bn_mean,
                   const float* __restrict__ bn_var,
                   const float* __restrict__ res_w,
                   const float* __restrict__ res_b,
                   const float* __restrict__ rbn_gamma,
                   const float* __restrict__ rbn_beta,
                   const float* __restrict__ rbn_mean,
                   const float* __restrict__ rbn_var,
                   float* __restrict__ out) {
  extern __shared__ unsigned char smem[];
  unsigned short* sw   = (unsigned short*)smem;
  float*          sf   = (float*)smem;
  unsigned short* wcat = sw + WCAT_US;
  unsigned short* awbT = sw + AWBT_US;
  float*          biasA  = sf + BIASA_F;
  float*          constc = sf + CONSTC_F;
  unsigned short* xb   = sw + XB_US;
  unsigned short* ybuf = sw + YB_US;
  unsigned short* rbuf = sw + RB_US;

  const int tid  = threadIdx.x;
  const int wave = tid >> 5;
  const int lane = tid & 31;
  const int half = lane >> 4;     // K-half selector per WMMA lane layout
  const int wl   = lane & 15;     // N / M-minor index

  const int n  = blockIdx.x / 25;
  const int tb = blockIdx.x % 25;
  const int t0 = tb * TT;

  // ------------------------------------------------------------------
  // Preamble: fold BN into weights/biases, stage A*edge^T, stage X tile
  // ------------------------------------------------------------------
  {
    // One thread per stacked-weight row (0..127 gcn, 128..255 res)
    const int row   = tid;
    const int c     = row & 127;
    const bool isres = row >= 128;
    const float g  = isres ? rbn_gamma[c] : bn_gamma[c];
    const float vv = isres ? rbn_var[c]   : bn_var[c];
    const float inv = g * rsqrtf(vv + 1e-5f);
    const vf4* wsrc = (const vf4*)((isres ? res_w : gcn_w) + c * CIN);
#pragma unroll 4
    for (int k4 = 0; k4 < CIN / 4; ++k4) {
      const vf4 wv = wsrc[k4];
      wcat[row * CIN + k4 * 4 + 0] = f2bf(wv.x * inv);
      wcat[row * CIN + k4 * 4 + 1] = f2bf(wv.y * inv);
      wcat[row * CIN + k4 * 4 + 2] = f2bf(wv.z * inv);
      wcat[row * CIN + k4 * 4 + 3] = f2bf(wv.w * inv);
    }
    biasA[row] = inv * (isres ? res_b[c] : gcn_b[c]);
    if (!isres) {
      const float rinv = rbn_gamma[c] * rsqrtf(rbn_var[c] + 1e-5f);
      constc[c] = (bn_beta[c] - bn_mean[c] * inv) +
                  (rbn_beta[c] - rbn_mean[c] * rinv);
    }
  }
  // (A * edge)^T into [w][v], padded 32x32 with zeros (zero K-pad = exact)
  for (int i = tid; i < 32 * 32; i += 256) {
    const int w = i >> 5, v = i & 31;
    unsigned short h = 0;
    if (v < VDIM && w < VDIM) h = f2bf(A[v * VDIM + w] * edge[v * VDIM + w]);
    awbT[w * 32 + v] = h;
  }
  // Zero ybuf (covers v=25..31 K-padding)
  {
    unsigned int* yz = (unsigned int*)ybuf;
    for (int i = tid; i < (COUT * YB_CSTRIDE) / 2; i += 256) yz[i] = 0u;
  }
  // X tile: global [k][300 contiguous (t,v)] -> LDS column-major bf16 [col][k]
  // x is single-use streaming data: non-temporal loads keep L2 clean.
  {
    const vf4* xbase = (const vf4*)(x + (size_t)(n * CIN) * (TDIM * VDIM) +
                                    (size_t)t0 * VDIM);
    for (int i = tid; i < CIN * 76; i += 256) {
      const int k = i / 76, j = i % 76;
      if (j < 75) {
        const vf4 xv = __builtin_nontemporal_load(
            xbase + (size_t)k * (TDIM * VDIM / 4) + j);
        const int col = j * 4;
        xb[(col + 0) * XB_STRIDE + k] = f2bf(xv.x);
        xb[(col + 1) * XB_STRIDE + k] = f2bf(xv.y);
        xb[(col + 2) * XB_STRIDE + k] = f2bf(xv.z);
        xb[(col + 3) * XB_STRIDE + k] = f2bf(xv.w);
      } else {
#pragma unroll
        for (int q = 0; q < 4; ++q) xb[(COLS + q) * XB_STRIDE + k] = 0;
      }
    }
  }
  __syncthreads();

  // ------------------------------------------------------------------
  // Stage A: Wcat[256x64] @ X[64x304]  (wave -> 2 row-tiles, loop cols)
  // ------------------------------------------------------------------
  {
    BF16Frag afr[2][2];
    float bval[2][8];
#pragma unroll
    for (int rt = 0; rt < 2; ++rt) {
      const int m = wave * 32 + rt * 16 + wl;
#pragma unroll
      for (int ks = 0; ks < 2; ++ks) {
        const int kb = ks * 32 + half * 8;   // A 16-bit lane layout
        afr[rt][ks].q[0] = *(const uint4*)&wcat[m * CIN + kb];
        afr[rt][ks].q[1] = *(const uint4*)&wcat[m * CIN + kb + 16];
      }
#pragma unroll
      for (int r = 0; r < 8; ++r)
        bval[rt][r] = biasA[wave * 32 + rt * 16 + r + half * 8];
    }
    // Wave-uniform destination select (branch-free epilogue):
    //   gcn rows -> ybuf[crow*392 + t*32 + v]
    //   res rows -> rbuf[(crow-128)*304 + col]
    const bool isres = (wave >= 4);
    unsigned short* dstbase = isres ? (rbuf - 128 * RB_CSTRIDE) : ybuf;
    const int cstride = isres ? RB_CSTRIDE : YB_CSTRIDE;

    for (int ct = 0; ct < NCT; ++ct) {
      const int col = ct * 16 + wl;
      BF16Frag bfr[2];
#pragma unroll
      for (int ks = 0; ks < 2; ++ks) {       // B 32x16 lane layout
        const int base = col * XB_STRIDE + ks * 32 + half * 8;
        bfr[ks].q[0] = *(const uint4*)&xb[base];
        bfr[ks].q[1] = *(const uint4*)&xb[base + 16];
      }
      const int t  = col / 25;
      const int vv = col - t * 25;
      const int colterm = isres ? col : (t * 32 + vv);
#pragma unroll
      for (int rt = 0; rt < 2; ++rt) {
        v8f acc = {};
        acc = wmma_bf16(afr[rt][0], bfr[0], acc);
        acc = wmma_bf16(afr[rt][1], bfr[1], acc);
        if (col < COLS) {
          const int cbase = wave * 32 + rt * 16 + half * 8;
#pragma unroll
          for (int r = 0; r < 8; ++r) {
            const unsigned short h = f2bf(acc[r] + bval[rt][r]);
            dstbase[(cbase + r) * cstride + colterm] = h;
          }
        }
      }
    }
  }
  __syncthreads();

  // ------------------------------------------------------------------
  // Stage B: per t, Y2[16c x 16w] = Y1[16c x 32v] @ AwT; fuse +const+res+relu
  // ------------------------------------------------------------------
  {
    const int c0 = wave * 16;
    BF16Frag bw[2];
#pragma unroll
    for (int wt = 0; wt < 2; ++wt) {
      const int w = wt * 16 + wl;
      bw[wt].q[0] = *(const uint4*)&awbT[w * 32 + half * 8];
      bw[wt].q[1] = *(const uint4*)&awbT[w * 32 + half * 8 + 16];
    }
    float cst[8];
#pragma unroll
    for (int r = 0; r < 8; ++r) cst[r] = constc[c0 + r + half * 8];

    for (int t = 0; t < TT; ++t) {
      BF16Frag af;
      {
        const int cA = c0 + wl;
        const int kb = half * 8;
        af.q[0] = *(const uint4*)&ybuf[cA * YB_CSTRIDE + t * 32 + kb];
        af.q[1] = *(const uint4*)&ybuf[cA * YB_CSTRIDE + t * 32 + kb + 16];
      }
      v8f accs[2];
      accs[0] = v8f{};
      accs[1] = v8f{};
      accs[0] = wmma_bf16(af, bw[0], accs[0]);
      accs[1] = wmma_bf16(af, bw[1], accs[1]);
#pragma unroll
      for (int wt = 0; wt < 2; ++wt) {
        const int w = wt * 16 + wl;
        if (w < VDIM) {
#pragma unroll
          for (int r = 0; r < 8; ++r) {
            const int c = c0 + r + half * 8;
            const float res = bf2f(rbuf[c * RB_CSTRIDE + t * VDIM + w]);
            float val = accs[wt][r] + cst[r] + res;
            val = fmaxf(val, 0.0f);
            // Write-once streaming output: non-temporal store.
            __builtin_nontemporal_store(
                val, &out[(((size_t)n * COUT + c) * TDIM + (t0 + t)) * VDIM + w]);
          }
        }
      }
    }
  }
}

extern "C" void kernel_launch(void* const* d_in, const int* in_sizes, int n_in,
                              void* d_out, int out_size, void* d_ws, size_t ws_size,
                              hipStream_t stream) {
  (void)in_sizes; (void)n_in; (void)d_ws; (void)ws_size; (void)out_size;
  const float* x         = (const float*)d_in[0];
  const float* A         = (const float*)d_in[1];
  const float* edge      = (const float*)d_in[2];
  const float* gcn_w     = (const float*)d_in[3];
  const float* gcn_b     = (const float*)d_in[4];
  const float* bn_gamma  = (const float*)d_in[5];
  const float* bn_beta   = (const float*)d_in[6];
  const float* bn_mean   = (const float*)d_in[7];
  const float* bn_var    = (const float*)d_in[8];
  const float* res_w     = (const float*)d_in[9];
  const float* res_b     = (const float*)d_in[10];
  const float* rbn_gamma = (const float*)d_in[11];
  const float* rbn_beta  = (const float*)d_in[12];
  const float* rbn_mean  = (const float*)d_in[13];
  const float* rbn_var   = (const float*)d_in[14];
  float* out = (float*)d_out;

  dim3 grid(NB * 25);   // 64 batches x 25 t-blocks of 12
  dim3 block(256);      // 8 waves (wave32)
  hipLaunchKernelGGL(stgcn_fused_kernel, grid, block, SMEM_BYTES, stream,
                     x, A, edge, gcn_w, gcn_b, bn_gamma, bn_beta, bn_mean,
                     bn_var, res_w, res_b, rbn_gamma, rbn_beta, rbn_mean,
                     rbn_var, out);
}
